// Projection1_40570261078386
// MI455X (gfx1250) — compile-verified, matrix operand order checked by reading, not run on hardware
//
#include <hip/hip_runtime.h>
#include <hip/hip_bf16.h>

typedef __attribute__((ext_vector_type(16))) __bf16 v16bf;
typedef __attribute__((ext_vector_type(8)))  __bf16 v8bf;
typedef __attribute__((ext_vector_type(8)))  float  v8f;
typedef __attribute__((ext_vector_type(4)))  float  v4f;

#define BATCH 16
#define SEQ   2048
#define DIM   256

static __device__ __forceinline__ v16bf combine8(v8bf a, v8bf b) {
    v16bf r;
#pragma unroll
    for (int i = 0; i < 8; ++i) { r[i] = a[i]; r[i + 8] = b[i]; }
    return r;
}

static __device__ __forceinline__ v8f wmma_bf16(v16bf a, v16bf b, v8f c) {
    return __builtin_amdgcn_wmma_f32_16x16x32_bf16(
        /*neg_a=*/false, a, /*neg_b=*/false, b,
        /*c_mod=*/(short)0, c, /*reuse_a=*/false, /*reuse_b=*/false);
}

// ---------------------------------------------------------------------------
// Kernel 0: split W (f32 [256,256]) into bf16 hi/lo parts.
// ---------------------------------------------------------------------------
__global__ void wsplit_kernel(const float* __restrict__ W,
                              __bf16* __restrict__ Wh, __bf16* __restrict__ Wl) {
    int i = blockIdx.x * blockDim.x + threadIdx.x;
    for (; i < DIM * DIM; i += gridDim.x * blockDim.x) {
        float v = W[i];
        __bf16 hi = (__bf16)v;
        Wh[i] = hi;
        Wl[i] = (__bf16)(v - (float)hi);
    }
}

// ---------------------------------------------------------------------------
// Kernel 1: projection p = x @ W^T + b, row-normalize, emit pn as bf16 hi/lo.
// One workgroup (8 waves) handles 16 rows x 256 cols; wave w owns col tiles
// 2w and 2w+1. bf16x3 split GEMM with f32 accumulation.
// ---------------------------------------------------------------------------
__global__ __launch_bounds__(256) void proj_kernel(
    const float* __restrict__ x, const __bf16* __restrict__ Wh,
    const __bf16* __restrict__ Wl, const float* __restrict__ bq,
    __bf16* __restrict__ pnh, __bf16* __restrict__ pnl) {
    __shared__ float ptile[16][260];   // padded to dodge bank conflicts
    __shared__ float norm2[16];

    const int tid  = threadIdx.x;
    const int lane = tid & 31;
    const int wave = tid >> 5;
    const int m    = lane & 15;       // column-within-tile / row-within-tile
    const int h    = lane >> 4;       // lane half selects K sub-pattern
    const int g0   = blockIdx.x * 16; // global row base (b*2048+n flattened)

    if (tid < 16) norm2[tid] = 0.0f;
    __syncthreads();

    const size_t xrow = (size_t)(g0 + m) * DIM;
    size_t wrow[2];
#pragma unroll
    for (int t = 0; t < 2; ++t)
        wrow[t] = (size_t)((2 * wave + t) * 16 + m) * DIM;

    v8f acc[2] = {};
#pragma unroll
    for (int k0 = 0; k0 < DIM; k0 += 32) {
        // A fragment: 16 f32 from this lane's x row, split to bf16 hi/lo.
        v16bf Ah, Al;
        {
            const size_t ka = xrow + k0 + 8 * h;
            v4f q0 = *(const v4f*)(x + ka);
            v4f q1 = *(const v4f*)(x + ka + 4);
            v4f q2 = *(const v4f*)(x + ka + 16);
            v4f q3 = *(const v4f*)(x + ka + 20);
            float vals[16] = {q0[0], q0[1], q0[2], q0[3], q1[0], q1[1], q1[2], q1[3],
                              q2[0], q2[1], q2[2], q2[3], q3[0], q3[1], q3[2], q3[3]};
#pragma unroll
            for (int i = 0; i < 16; ++i) {
                __bf16 hi = (__bf16)vals[i];
                Ah[i] = hi;
                Al[i] = (__bf16)(vals[i] - (float)hi);
            }
        }
#pragma unroll
        for (int t = 0; t < 2; ++t) {
            const size_t kb = wrow[t] + k0 + 8 * h;
            v16bf Bh = combine8(*(const v8bf*)(Wh + kb), *(const v8bf*)(Wh + kb + 16));
            v16bf Bl = combine8(*(const v8bf*)(Wl + kb), *(const v8bf*)(Wl + kb + 16));
            acc[t] = wmma_bf16(Ah, Bh, acc[t]);
            acc[t] = wmma_bf16(Ah, Bl, acc[t]);
            acc[t] = wmma_bf16(Al, Bh, acc[t]);
        }
    }

    // Bias add, stash raw p into LDS, accumulate row sums of squares.
#pragma unroll
    for (int t = 0; t < 2; ++t) {
        const int ccol = (2 * wave + t) * 16 + m;
        const float bv = bq[ccol];
#pragma unroll
        for (int r = 0; r < 8; ++r) {
            float v = acc[t][r] + bv;
            acc[t][r] = v;
            ptile[r + 8 * h][ccol] = v;
        }
    }
#pragma unroll
    for (int r = 0; r < 8; ++r) {
        float t2 = acc[0][r] * acc[0][r] + acc[1][r] * acc[1][r];
        t2 += __shfl_xor(t2, 1, 32);
        t2 += __shfl_xor(t2, 2, 32);
        t2 += __shfl_xor(t2, 4, 32);
        t2 += __shfl_xor(t2, 8, 32);   // stays within each 16-lane half
        if (m == 0) atomicAdd(&norm2[r + 8 * h], t2);
    }
    __syncthreads();

    // Normalize and emit bf16 hi/lo, coalesced 16B stores.
    const int row = tid >> 4;
    const int seg = tid & 15;
    const float rn = rsqrtf(norm2[row]);
    v8bf oh[2], ol[2];
#pragma unroll
    for (int q = 0; q < 2; ++q) {
#pragma unroll
        for (int i = 0; i < 8; ++i) {
            float v = ptile[row][seg * 16 + q * 8 + i] * rn;
            __bf16 hi = (__bf16)v;
            oh[q][i] = hi;
            ol[q][i] = (__bf16)(v - (float)hi);
        }
    }
    const size_t o = (size_t)(g0 + row) * DIM + seg * 16;
    *(v8bf*)(pnh + o)     = oh[0];
    *(v8bf*)(pnh + o + 8) = oh[1];
    *(v8bf*)(pnl + o)     = ol[0];
    *(v8bf*)(pnl + o + 8) = ol[1];
}

// ---------------------------------------------------------------------------
// Kernel 2: Gram matrix attn[b] = pn[b] @ pn[b]^T via bf16x3 WMMA.
// 8 waves per WG arranged 4x2; each wave owns a 32x32 output block
// (2x2 tiles of 16x16). Grid: (2048/64, 2048/128, 16).
// ---------------------------------------------------------------------------
__global__ __launch_bounds__(256) void gram_kernel(
    const __bf16* __restrict__ pnh, const __bf16* __restrict__ pnl,
    float* __restrict__ out) {
    const int lane = threadIdx.x & 31;
    const int wave = threadIdx.x >> 5;
    const int m = lane & 15, h = lane >> 4;
    const int wi = wave >> 1, wj = wave & 1;
    const size_t zrow = (size_t)blockIdx.z * SEQ;
    const int i0 = blockIdx.y * 128 + wi * 32;
    const int j0 = blockIdx.x * 64 + wj * 32;

    size_t aoff[2], boff[2];
#pragma unroll
    for (int t = 0; t < 2; ++t) {
        aoff[t] = (zrow + i0 + t * 16 + m) * DIM;
        boff[t] = (zrow + j0 + t * 16 + m) * DIM;
    }

    v8f acc[2][2] = {};
#pragma unroll
    for (int k0 = 0; k0 < DIM; k0 += 32) {
        if (k0 + 32 < DIM) {  // warm next K tiles (global_prefetch_b8)
            __builtin_prefetch(pnh + aoff[0] + k0 + 32 + 8 * h, 0, 0);
            __builtin_prefetch(pnh + boff[0] + k0 + 32 + 8 * h, 0, 0);
        }
        v16bf Ah[2], Al[2], Bh[2], Bl[2];
#pragma unroll
        for (int t = 0; t < 2; ++t) {
            const size_t ka = aoff[t] + k0 + 8 * h;
            Ah[t] = combine8(*(const v8bf*)(pnh + ka), *(const v8bf*)(pnh + ka + 16));
            Al[t] = combine8(*(const v8bf*)(pnl + ka), *(const v8bf*)(pnl + ka + 16));
            const size_t kb = boff[t] + k0 + 8 * h;
            Bh[t] = combine8(*(const v8bf*)(pnh + kb), *(const v8bf*)(pnh + kb + 16));
            Bl[t] = combine8(*(const v8bf*)(pnl + kb), *(const v8bf*)(pnl + kb + 16));
        }
#pragma unroll
        for (int ti = 0; ti < 2; ++ti)
#pragma unroll
            for (int tj = 0; tj < 2; ++tj) {
                acc[ti][tj] = wmma_bf16(Ah[ti], Bh[tj], acc[ti][tj]);
                acc[ti][tj] = wmma_bf16(Ah[ti], Bl[tj], acc[ti][tj]);
                acc[ti][tj] = wmma_bf16(Al[ti], Bh[tj], acc[ti][tj]);
            }
    }

#pragma unroll
    for (int ti = 0; ti < 2; ++ti)
#pragma unroll
        for (int tj = 0; tj < 2; ++tj) {
            const int jcol = j0 + tj * 16 + m;
#pragma unroll
            for (int r = 0; r < 8; ++r) {
                const int irow = i0 + ti * 16 + r + 8 * h;
                out[(zrow + irow) * SEQ + jcol] = acc[ti][tj][r];
            }
        }
}

extern "C" void kernel_launch(void* const* d_in, const int* in_sizes, int n_in,
                              void* d_out, int out_size, void* d_ws, size_t ws_size,
                              hipStream_t stream) {
    const float* x  = (const float*)d_in[0];   // [16, 2048, 256]
    const float* Wq = (const float*)d_in[1];   // [256, 256]
    const float* bq = (const float*)d_in[2];   // [256]
    float* out = (float*)d_out;                // [16, 2048, 2048]

    char* ws = (char*)d_ws;
    __bf16* pnh = (__bf16*)ws;                                      // 16 MiB
    __bf16* pnl = (__bf16*)(ws + (size_t)16 * 1024 * 1024);         // 16 MiB
    __bf16* Wh  = (__bf16*)(ws + (size_t)32 * 1024 * 1024);         // 128 KiB
    __bf16* Wl  = (__bf16*)(ws + (size_t)32 * 1024 * 1024 + 131072);// 128 KiB

    wsplit_kernel<<<64, 256, 0, stream>>>(Wq, Wh, Wl);
    proj_kernel<<<(BATCH * SEQ) / 16, 256, 0, stream>>>(x, Wh, Wl, bq, pnh, pnl);
    dim3 grid(SEQ / 64, SEQ / 128, BATCH);
    gram_kernel<<<grid, 256, 0, stream>>>(pnh, pnl, out);
}
